// ScaledDotProductAttention_13649406066811
// MI455X (gfx1250) — matrix operand order repaired and checked by reading, not compiled, not run on hardware
//
#include <hip/hip_runtime.h>
#include <stdint.h>

typedef __attribute__((ext_vector_type(16))) __bf16 v16bf;
typedef __attribute__((ext_vector_type(8)))  __bf16 v8bf;
typedef __attribute__((ext_vector_type(2)))  __bf16 v2bf;
typedef __attribute__((ext_vector_type(8)))  float  v8f;

static constexpr int S_LEN = 2048;   // sequence length
static constexpr int DH    = 64;     // head dim
static constexpr int WAVES = 8;      // waves per block (wave32) -> 128 q-rows/block
static constexpr int KCH   = 32;     // keys per chunk (== WMMA K for bf16)
static constexpr int STG   = 68;     // padded f32 stage row stride (272B)
static constexpr int KTS   = 72;     // padded bf16 K-tile row stride (144B)
static constexpr int VTS   = 40;     // padded bf16 V^T-tile row stride (80B)
// Q pre-scale: 1/temperature folded with log2(e) so softmax runs in log2 domain.
static constexpr float QSCALE = 0.125f * 1.44269504088896340736f;

// Async copy: 16 bytes per lane, global -> LDS, tracked by ASYNCcnt (CDNA5).
__device__ __forceinline__ void async_cp_b128(const float* gp, const float* lp) {
  unsigned loff = (unsigned)(uintptr_t)lp;   // LDS aperture: addr[31:0] == LDS byte offset
  asm volatile("global_load_async_to_lds_b128 %0, %1, off"
               :: "v"(loff), "v"(gp) : "memory");
}

// Half-wave exchange (lanes L <-> L^16) via ds_swizzle SWAPX16.
__device__ __forceinline__ float swap16(float x) {
  return __int_as_float(__builtin_amdgcn_ds_swizzle(__float_as_int(x), 0x401f));
}

// 16 contiguous bf16 in LDS -> one WMMA fragment register.
__device__ __forceinline__ v16bf ld16(const __bf16* p) {
  v8bf lo = *(const v8bf*)(p);
  v8bf hi = *(const v8bf*)(p + 8);
  v16bf r;
#pragma unroll
  for (int i = 0; i < 8; ++i) { r[i] = lo[i]; r[i + 8] = hi[i]; }
  return r;
}

__global__ __launch_bounds__(256) void fa_fwd_bf16wmma(
    const float* __restrict__ qg, const float* __restrict__ kg,
    const float* __restrict__ vg, float* __restrict__ og) {
  // Double-buffered f32 staging (filled by async-to-LDS DMA).
  __shared__ __align__(16) float  st_k[2][KCH * STG];
  __shared__ __align__(16) float  st_v[2][KCH * STG];
  // Shared bf16 operand tiles (converted once per block per chunk).
  __shared__ __align__(16) __bf16 kt[KCH * KTS];   // K chunk, row-major [key][d]
  __shared__ __align__(16) __bf16 vtt[DH * VTS];   // V chunk, transposed [d][key]
  // Per-wave softmax staging.
  __shared__ __align__(16) float  s_sc[WAVES][16 * KCH];
  __shared__ __align__(16) __bf16 s_p[WAVES][16 * KCH];
  __shared__ __align__(16) float  s_stat[WAVES][16];

  const int lane = threadIdx.x & 31;
  const int wv   = threadIdx.x >> 5;
  const int b    = blockIdx.y;
  const int q0   = blockIdx.x * (16 * WAVES) + wv * 16;

  const float* qb = qg + ((size_t)b * S_LEN + q0) * DH;
  const float* kb = kg + (size_t)b * S_LEN * DH;
  const float* vb = vg + (size_t)b * S_LEN * DH;

  const int nlo   = lane & 15;   // A-layout row M; B/C-layout column N
  const int hi    = lane >> 4;   // half-wave select
  const int rbase = hi * 8;      // C-layout row base
  const int akoff = hi * 8;      // A-layout K offset {0,8}
  const int bkoff = hi * 16;     // K offset {0,16}: contiguous k order, used for A and B

  // ---- Q tile as two bf16 A-fragments, pre-scaled by (1/T)*log2(e) ----
  const float* qrow = qb + (size_t)nlo * DH;
  v16bf qa0, qa1;
#pragma unroll
  for (int i = 0; i < 8; ++i) {
    qa0[i]     = (__bf16)(qrow[bkoff + i] * QSCALE);
    qa0[i + 8] = (__bf16)(qrow[bkoff + 8 + i] * QSCALE);
    qa1[i]     = (__bf16)(qrow[32 + bkoff + i] * QSCALE);
    qa1[i + 8] = (__bf16)(qrow[32 + bkoff + 8 + i] * QSCALE);
  }

  v8f acc[4] = {};
  float m_row = -__builtin_inff();   // log2-domain running row max
  float l_row = 0.0f;                // running row sum
  const v8f zero_c = {};

  // One chunk's K+V async copies: 4 x b128 per wave; 8 waves cover 16KB.
  auto issue_chunk = [&](int s0, int buf) {
#pragma unroll
    for (int i = 0; i < 2; ++i) {
      const int e   = ((wv * 2 + i) * 32 + lane) * 4;  // flat f32 index in 32x64 chunk
      const int key = e >> 6;
      const int d   = e & 63;
      async_cp_b128(kb + (size_t)(s0 + key) * DH + d, &st_k[buf][key * STG + d]);
      async_cp_b128(vb + (size_t)(s0 + key) * DH + d, &st_v[buf][key * STG + d]);
    }
  };

  // Cooperative f32 -> bf16 conversion into the shared operand tiles (256 thr).
  auto convert_chunk = [&](int buf) {
    const int t = threadIdx.x;
    // K: thread -> (key, 8-wide d group); one b128 store.
    {
      const int key = t >> 3, dg = (t & 7) * 8;
      const float* pk = &st_k[buf][key * STG + dg];
      v8bf a;
#pragma unroll
      for (int i = 0; i < 8; ++i) a[i] = (__bf16)pk[i];
      *(v8bf*)&kt[key * KTS + dg] = a;
    }
    // V (transposed): thread -> (key pair, 4-wide d group); 4 x b32 stores.
    {
      const int key0 = (t >> 4) * 2, dg2 = (t & 15) * 4;
      const float* pv0 = &st_v[buf][key0 * STG + dg2];
      const float* pv1 = &st_v[buf][(key0 + 1) * STG + dg2];
#pragma unroll
      for (int i = 0; i < 4; ++i) {
        v2bf pp; pp[0] = (__bf16)pv0[i]; pp[1] = (__bf16)pv1[i];
        *(v2bf*)&vtt[(dg2 + i) * VTS + key0] = pp;
      }
    }
  };

  issue_chunk(0, 0);

  for (int c = 0; c < S_LEN / KCH; ++c) {
    const int s0 = c * KCH;

    // Prefetch next chunk into the other buffer; retire only the older group.
    if (s0 + KCH < S_LEN) {
      issue_chunk(s0 + KCH, (c + 1) & 1);
      asm volatile("s_wait_asynccnt 0x4" ::: "memory");
    } else {
      asm volatile("s_wait_asynccnt 0x0" ::: "memory");
    }
    __syncthreads();               // all waves' async groups for chunk c retired
    convert_chunk(c & 1);
    __syncthreads();               // bf16 tiles ready for all waves

    // ---- scores = (Q*scale) @ K^T: two 16x16 f32 C tiles, K over D=64 ----
#pragma unroll
    for (int cg = 0; cg < 2; ++cg) {
      const __bf16* kp = &kt[(cg * 16 + nlo) * KTS + bkoff];
      v16bf kf0 = ld16(kp);        // D 0..31 half for this half-wave
      v16bf kf1 = ld16(kp + 32);   // D 32..63 half
      v8f sc = __builtin_amdgcn_wmma_f32_16x16x32_bf16(
          false, qa0, false, kf0, (short)0, zero_c, false, false);
      sc = __builtin_amdgcn_wmma_f32_16x16x32_bf16(
          false, qa1, false, kf1, (short)0, sc, false, false);
#pragma unroll
      for (int r = 0; r < 8; ++r)
        s_sc[wv][(rbase + r) * KCH + cg * 16 + nlo] = sc[r];
    }
    asm volatile("s_wait_dscnt 0x0" ::: "memory");

    // ---- online softmax (log2 domain), all 32 lanes active:
    //      lane L owns row L%16, columns [hi*16, hi*16+16); halves combined
    //      with ds_swizzle SWAPX16. ----
    {
      const float* srow = &s_sc[wv][nlo * KCH + hi * 16];
      float cmax = srow[0];
#pragma unroll
      for (int j = 1; j < 16; ++j) cmax = fmaxf(cmax, srow[j]);
      cmax = fmaxf(cmax, swap16(cmax));            // full-row max
      float m_new = fmaxf(m_row, cmax);
      float alpha = __builtin_amdgcn_exp2f(m_row - m_new);
      float sum = 0.0f;
#pragma unroll
      for (int j = 0; j < 16; j += 2) {
        float p0 = __builtin_amdgcn_exp2f(srow[j] - m_new);
        float p1 = __builtin_amdgcn_exp2f(srow[j + 1] - m_new);
        sum += p0 + p1;
        v2bf pp; pp[0] = (__bf16)p0; pp[1] = (__bf16)p1;
        *(v2bf*)&s_p[wv][nlo * KCH + hi * 16 + j] = pp;
      }
      sum += swap16(sum);                          // full-row sum
      l_row = l_row * alpha + sum;
      m_row = m_new;
      if (lane < 16) s_stat[wv][lane] = alpha;
    }
    asm volatile("s_wait_dscnt 0x0" ::: "memory");

    // ---- P as one bf16 A-fragment ----
    const __bf16* prow = &s_p[wv][nlo * KCH + bkoff];
    v8bf p_lo = *(const v8bf*)(prow);
    v8bf p_hi = *(const v8bf*)(prow + 8);
    v16bf pf;
#pragma unroll
    for (int i = 0; i < 8; ++i) { pf[i] = p_lo[i]; pf[i + 8] = p_hi[i]; }

    // ---- rescale O by alpha, then O += P @ V (V from transposed LDS tile) ----
    float alr[8];
#pragma unroll
    for (int r = 0; r < 8; ++r) alr[r] = s_stat[wv][rbase + r];
#pragma unroll
    for (int t = 0; t < 4; ++t) {
#pragma unroll
      for (int r = 0; r < 8; ++r) acc[t][r] *= alr[r];
    }
#pragma unroll
    for (int t = 0; t < 4; ++t) {
      v16bf vf = ld16(&vtt[(t * 16 + nlo) * VTS + bkoff]);
      acc[t] = __builtin_amdgcn_wmma_f32_16x16x32_bf16(
          false, pf, false, vf, (short)0, acc[t], false, false);
    }
  }

  // ---- final normalization and store ----
  if (lane < 16) s_stat[wv][lane] = 1.0f / l_row;
  asm volatile("s_wait_dscnt 0x0" ::: "memory");
  float li[8];
#pragma unroll
  for (int r = 0; r < 8; ++r) li[r] = s_stat[wv][rbase + r];

  float* ob = og + ((size_t)b * S_LEN + q0) * DH;
#pragma unroll
  for (int t = 0; t < 4; ++t) {
#pragma unroll
    for (int r = 0; r < 8; ++r)
      ob[(size_t)(rbase + r) * DH + t * 16 + nlo] = acc[t][r] * li[r];
  }
}

extern "C" void kernel_launch(void* const* d_in, const int* in_sizes, int n_in,
                              void* d_out, int out_size, void* d_ws, size_t ws_size,
                              hipStream_t stream) {
  (void)n_in; (void)out_size; (void)d_ws; (void)ws_size;
  const float* q = (const float*)d_in[0];
  const float* k = (const float*)d_in[1];
  const float* v = (const float*)d_in[2];
  float* out = (float*)d_out;

  const int B = in_sizes[0] / (S_LEN * DH);   // 32 for the reference shapes
  dim3 grid(S_LEN / (16 * WAVES), B);
  dim3 block(32 * WAVES);
  fa_fwd_bf16wmma<<<grid, block, 0, stream>>>(q, k, v, out);
}
